// CopyGenerator_hype_52432960749620
// MI455X (gfx1250) — matrix operand order repaired and checked by reading, not compiled, not run on hardware
//
#include <hip/hip_runtime.h>
#include <hip/hip_bf16.h>
#include <math.h>

typedef __attribute__((ext_vector_type(16))) __bf16 v16bf;
typedef __attribute__((ext_vector_type(8)))  float  v8f;

#define N_ROWS 2048
#define DIM    1024
#define VOCAB  32000
#define SLEN   512
#define BATCH  32
#define CVOC   500
#define TLEN   64
#define OUTC   32500   // VOCAB + CVOC
#define MAXN   0.996f  // (1 - 4e-3)/sqrt(c), c=1

// ---------------------------------------------------------------- converts
__global__ void k_f32_to_bf16(const float* __restrict__ in,
                              unsigned short* __restrict__ out, int n) {
  int i = blockIdx.x * blockDim.x + threadIdx.x;
  int stride = gridDim.x * blockDim.x;
  for (; i < n; i += stride) {
    __hip_bfloat16 h = __float2bfloat16(in[i]);
    out[i] = *reinterpret_cast<unsigned short*>(&h);
  }
}

// ---------------------------------------------------------------- sum(b^2)
__global__ void k_b2sum(const float* __restrict__ b, float* __restrict__ b2) {
  __shared__ float buf[256];
  int tid = threadIdx.x;
  float s = 0.f;
  for (int v = tid; v < VOCAB; v += 256) { float x = b[v]; s += x * x; }
  buf[tid] = s; __syncthreads();
  for (int off = 128; off > 0; off >>= 1) {
    if (tid < off) buf[tid] += buf[tid + off];
    __syncthreads();
  }
  if (tid == 0) *b2 = buf[0];
}

// ---------------------------------------------------------------- WMMA GEMM
// Mx[m,v] = sum_k hidden[m,k] * W[v,k] ; block tile 128x128, 8 waves,
// wave tile 64(M)x32(N) = 4x2 of 16x16x32 bf16 WMMA, K loop over 1024,
// fully unrolled + double-buffered fragments: all loads are base+imm-offset,
// prefetch of K+32/K+64 overlaps current WMMAs (partial s_wait_loadcnt).
__device__ __forceinline__ v16bf ld_frag2(const unsigned short* p, int off2) {
  union { uint4 u[2]; v16bf v; } x;
  x.u[0] = *reinterpret_cast<const uint4*>(p);
  x.u[1] = *reinterpret_cast<const uint4*>(p + off2);
  return x.v;
}

__global__ __launch_bounds__(256)
void k_gemm_wmma(const unsigned short* __restrict__ Abf,   // hidden bf16 [2048][1024]
                 const unsigned short* __restrict__ Bbf,   // W bf16 [32000][1024]
                 float* __restrict__ out) {                // [2048][32500]
  const int tid  = threadIdx.x;
  const int lane = tid & 31;
  const int wave = tid >> 5;
  const int wm = wave >> 2, wn = wave & 3;
  const int ml = lane & 15, g = lane >> 4;
  const int m0 = blockIdx.y * 128 + wm * 64;
  const int n0 = blockIdx.x * 128 + wn * 32;

  const v8f vzero = {0.f,0.f,0.f,0.f,0.f,0.f,0.f,0.f};
  v8f acc[4][2];
#pragma unroll
  for (int i = 0; i < 4; ++i) { acc[i][0] = vzero; acc[i][1] = vzero; }

  const unsigned short* arow[4];
#pragma unroll
  for (int mi = 0; mi < 4; ++mi)
    arow[mi] = Abf + (size_t)(m0 + mi * 16 + ml) * DIM + g * 8;   // K = g*8 + i, +16
  const unsigned short* brow[2];
#pragma unroll
  for (int ni = 0; ni < 2; ++ni)
    brow[ni] = Bbf + (size_t)(n0 + ni * 16 + ml) * DIM + g * 8;   // K = g*8 + i, +16

  v16bf bf0[2], af0[4], bf1[2], af1[4];
#pragma unroll
  for (int ni = 0; ni < 2; ++ni) bf0[ni] = ld_frag2(brow[ni], 16);
#pragma unroll
  for (int mi = 0; mi < 4; ++mi) af0[mi] = ld_frag2(arow[mi], 16);

#pragma unroll
  for (int kk = 0; kk < DIM; kk += 64) {
    // prefetch K+32 into buf1 (always in range)
#pragma unroll
    for (int ni = 0; ni < 2; ++ni) bf1[ni] = ld_frag2(brow[ni] + kk + 32, 16);
#pragma unroll
    for (int mi = 0; mi < 4; ++mi) af1[mi] = ld_frag2(arow[mi] + kk + 32, 16);
    // compute on buf0 (K = kk)
#pragma unroll
    for (int mi = 0; mi < 4; ++mi) {
      acc[mi][0] = __builtin_amdgcn_wmma_f32_16x16x32_bf16(
          false, af0[mi], false, bf0[0], (short)0, acc[mi][0], false, false);
      acc[mi][1] = __builtin_amdgcn_wmma_f32_16x16x32_bf16(
          false, af0[mi], false, bf0[1], (short)0, acc[mi][1], false, false);
    }
    // prefetch K+64 into buf0 (statically dead on the final unrolled iteration)
    if (kk + 64 < DIM) {
#pragma unroll
      for (int ni = 0; ni < 2; ++ni) bf0[ni] = ld_frag2(brow[ni] + kk + 64, 16);
#pragma unroll
      for (int mi = 0; mi < 4; ++mi) af0[mi] = ld_frag2(arow[mi] + kk + 64, 16);
    }
    // compute on buf1 (K = kk + 32)
#pragma unroll
    for (int mi = 0; mi < 4; ++mi) {
      acc[mi][0] = __builtin_amdgcn_wmma_f32_16x16x32_bf16(
          false, af1[mi], false, bf1[0], (short)0, acc[mi][0], false, false);
      acc[mi][1] = __builtin_amdgcn_wmma_f32_16x16x32_bf16(
          false, af1[mi], false, bf1[1], (short)0, acc[mi][1], false, false);
    }
  }

#pragma unroll
  for (int mi = 0; mi < 4; ++mi)
#pragma unroll
    for (int ni = 0; ni < 2; ++ni) {
      const int col = n0 + ni * 16 + ml;
      const size_t rbase = (size_t)(m0 + mi * 16 + g * 8);
#pragma unroll
      for (int r = 0; r < 8; ++r)
        out[(rbase + r) * OUTC + col] = acc[mi][ni][r];
    }
}

// ---------------------------------------------------------------- per-row sums over Mx
__global__ void k_rowred(const float* __restrict__ out, const float* __restrict__ b,
                         float* __restrict__ row_s2, float* __restrict__ row_sb) {
  const int r = blockIdx.x, tid = threadIdx.x;
  const float* row = out + (size_t)r * OUTC;
  float s2 = 0.f, sb = 0.f;
  for (int v = tid; v < VOCAB; v += 256) {
    float x = row[v];
    s2 += x * x;
    sb += x * b[v];
  }
  __shared__ float b0[256], b1[256];
  b0[tid] = s2; b1[tid] = sb; __syncthreads();
  for (int off = 128; off > 0; off >>= 1) {
    if (tid < off) { b0[tid] += b0[tid + off]; b1[tid] += b1[tid + off]; }
    __syncthreads();
  }
  if (tid == 0) { row_s2[r] = b0[0]; row_sb[r] = b1[0]; }
}

// ---------------------------------------------------------------- per-row P/Q and p_copy
__device__ __forceinline__ float artanh_clip(float x) {
  x = fminf(fmaxf(x, -1.f + 1e-7f), 1.f - 1e-7f);
  return 0.5f * logf((1.f + x) / (1.f - x));
}

__global__ void k_scalars(const float* __restrict__ hidden, const float* __restrict__ Wc,
                          const float* __restrict__ bcp, const float* __restrict__ b2p,
                          const float* __restrict__ row_s2, const float* __restrict__ row_sb,
                          float* __restrict__ P, float* __restrict__ Q,
                          float* __restrict__ omp, float* __restrict__ pcp) {
  const int r = blockIdx.x, tid = threadIdx.x;
  const float* hr = hidden + (size_t)r * DIM;
  float sx2 = 0.f, mc = 0.f;
  for (int k = tid; k < DIM; k += 256) {
    float h = hr[k];
    sx2 += h * h;
    mc  += h * Wc[k];
  }
  __shared__ float b0[256], b1[256];
  b0[tid] = sx2; b1[tid] = mc; __syncthreads();
  for (int off = 128; off > 0; off >>= 1) {
    if (tid < off) { b0[tid] += b0[tid + off]; b1[tid] += b1[tid + off]; }
    __syncthreads();
  }
  if (tid != 0) return;
  sx2 = b0[0]; mc = b1[0];

  const float xn  = fmaxf(sqrtf(sx2), 1e-15f);
  const float art = artanh_clip(xn);          // sqrt(c)=1
  // ---- main vocab path: logits = P*Mx + Q*b
  {
    float S2  = row_s2[r], Sb = row_sb[r], y2 = *b2p;
    float Mxn = fmaxf(sqrtf(S2), 1e-15f);
    float t   = tanhf(Mxn / xn * art);
    float scale = t / Mxn;                    // out = scale*Mx
    float x2 = scale * scale * S2;
    float xy = scale * Sb;
    float alpha = 1.f + 2.f * xy + y2;
    float beta  = 1.f - x2;
    float den   = fmaxf(1.f + 2.f * xy + x2 * y2, 1e-15f);
    float P0 = alpha * scale / den;
    float Q0 = beta / den;
    float n2 = (alpha * alpha * x2 + 2.f * alpha * beta * xy + beta * beta * y2) / (den * den);
    float nrm = fmaxf(sqrtf(fmaxf(n2, 0.f)), 1e-15f);
    float sf = (nrm > MAXN) ? (MAXN / nrm) : 1.f;
    P[r] = sf * P0;
    Q[r] = sf * Q0;
  }
  // ---- p_copy path (V = 1)
  {
    float bcv = *bcp;
    float mxn = fmaxf(fabsf(mc), 1e-15f);
    float t   = tanhf(mxn / xn * art);
    float sc_ = t / mxn;
    float oc  = sc_ * mc;
    float x2 = oc * oc, y2 = bcv * bcv, xy = oc * bcv;
    float alpha = 1.f + 2.f * xy + y2;
    float beta  = 1.f - x2;
    float den   = fmaxf(1.f + 2.f * xy + x2 * y2, 1e-15f);
    float val = (alpha * oc + beta * bcv) / den;
    float nc = fmaxf(fabsf(val), 1e-15f);
    if (nc > MAXN) val *= MAXN / nc;
    float pc = 1.f / (1.f + __expf(-val));
    pcp[r] = pc;
    omp[r] = 1.f - pc;
  }
}

// ---------------------------------------------------------------- online softmax stats
__global__ void k_softstats(const float* __restrict__ out, const float* __restrict__ b,
                            const float* __restrict__ P, const float* __restrict__ Q,
                            const int* __restrict__ padp,
                            float* __restrict__ rm, float* __restrict__ rs) {
  const int r = blockIdx.x, tid = threadIdx.x;
  const float* row = out + (size_t)r * OUTC;
  const float p = P[r], q = Q[r];
  const int pad = *padp;
  float m = -INFINITY, s = 0.f;
  for (int v = tid; v < VOCAB; v += 256) {
    if (v == pad) continue;
    float l = p * row[v] + q * b[v];
    float nm = fmaxf(m, l);
    s = s * __expf(m - nm) + __expf(l - nm);
    m = nm;
  }
  __shared__ float sm[256], ss[256];
  sm[tid] = m; ss[tid] = s; __syncthreads();
  for (int off = 128; off > 0; off >>= 1) {
    if (tid < off) {
      float m1 = sm[tid], s1 = ss[tid];
      float m2 = sm[tid + off], s2 = ss[tid + off];
      float M = fmaxf(m1, m2);
      float S = (M == -INFINITY) ? 0.f : (s1 * __expf(m1 - M) + s2 * __expf(m2 - M));
      sm[tid] = M; ss[tid] = S;
    }
    __syncthreads();
  }
  if (tid == 0) { rm[r] = sm[0]; rs[r] = ss[0]; }
}

// ---------------------------------------------------------------- write probabilities
__global__ void k_writeprob(float* __restrict__ out, const float* __restrict__ b,
                            const float* __restrict__ P, const float* __restrict__ Q,
                            const float* __restrict__ omp,
                            const float* __restrict__ rm, const float* __restrict__ rs,
                            const int* __restrict__ padp) {
  const int r = blockIdx.y;
  const int v = blockIdx.x * 256 + threadIdx.x;   // 125*256 == 32000
  float* row = out + (size_t)r * OUTC;
  float val = 0.f;
  if (v != *padp) {
    float l = P[r] * row[v] + Q[r] * b[v];
    val = __expf(l - rm[r]) / rs[r] * omp[r];
  }
  row[v] = val;
}

// ---------------------------------------------------------------- copy attention einsum
// copy[t,b,c] = sum_s attn[t*32+b, s] * p_copy[t*32+b] * src_map[s,b,c]
// block: fixed b (blockIdx.y), c-tile of 64 (blockIdx.x), full 64 t-rows, K=512
__global__ __launch_bounds__(256)
void k_copygen(const float* __restrict__ attn, const float* __restrict__ src,
               const float* __restrict__ pcp, float* __restrict__ out) {
  const int bb = blockIdx.y;
  const int c0 = blockIdx.x * 64;
  const int tid = threadIdx.x;
  const int tx = tid & 15, ty = tid >> 4;

  __shared__ float As[64][33];   // [t][s]
  __shared__ float Bs[32][65];   // [s][c]

  float acc[4][4];
#pragma unroll
  for (int i = 0; i < 4; ++i)
#pragma unroll
    for (int j = 0; j < 4; ++j) acc[i][j] = 0.f;

  for (int s0 = 0; s0 < SLEN; s0 += 32) {
    for (int i = tid; i < 64 * 32; i += 256) {
      int tt = i >> 5, ss2 = i & 31;
      int rw = tt * BATCH + bb;
      As[tt][ss2] = attn[(size_t)rw * SLEN + s0 + ss2] * pcp[rw];
    }
    for (int i = tid; i < 32 * 64; i += 256) {
      int ss2 = i >> 6, cc = i & 63;
      int c = c0 + cc;
      Bs[ss2][cc] = (c < CVOC)
          ? src[(size_t)(s0 + ss2) * (BATCH * CVOC) + bb * CVOC + c] : 0.f;
    }
    __syncthreads();
#pragma unroll
    for (int ss2 = 0; ss2 < 32; ++ss2) {
      float ar[4], bc[4];
#pragma unroll
      for (int i = 0; i < 4; ++i) ar[i] = As[ty * 4 + i][ss2];
#pragma unroll
      for (int j = 0; j < 4; ++j) bc[j] = Bs[ss2][tx * 4 + j];
#pragma unroll
      for (int i = 0; i < 4; ++i)
#pragma unroll
        for (int j = 0; j < 4; ++j) acc[i][j] = fmaf(ar[i], bc[j], acc[i][j]);
    }
    __syncthreads();
  }
#pragma unroll
  for (int i = 0; i < 4; ++i) {
    int rw = (ty * 4 + i) * BATCH + bb;
#pragma unroll
    for (int j = 0; j < 4; ++j) {
      int c = c0 + tx * 4 + j;
      if (c < CVOC) out[(size_t)rw * OUTC + VOCAB + c] = acc[i][j];
    }
  }
}

// ---------------------------------------------------------------- launch
extern "C" void kernel_launch(void* const* d_in, const int* in_sizes, int n_in,
                              void* d_out, int out_size, void* d_ws, size_t ws_size,
                              hipStream_t stream) {
  (void)in_sizes; (void)n_in; (void)out_size; (void)ws_size;
  const float* hidden = (const float*)d_in[0];  // [2048][1024]
  const float* attn   = (const float*)d_in[1];  // [2048][512]
  const float* src    = (const float*)d_in[2];  // [512][32][500]
  const float* W      = (const float*)d_in[3];  // [32000][1024]
  const float* b      = (const float*)d_in[4];  // [32000]
  const float* Wc     = (const float*)d_in[5];  // [1024]
  const float* bc     = (const float*)d_in[6];  // [1]
  const int*   pad    = (const int*)d_in[7];    // [1]
  float* out = (float*)d_out;                   // [2048][32500]

  size_t off = 0;
  auto alloc = [&](size_t bytes) {
    void* p = (char*)d_ws + off;
    off += (bytes + 255) & ~(size_t)255;
    return p;
  };
  unsigned short* hidB = (unsigned short*)alloc((size_t)N_ROWS * DIM * 2);
  unsigned short* WB   = (unsigned short*)alloc((size_t)VOCAB * DIM * 2);
  float* row_s2 = (float*)alloc(N_ROWS * 4);
  float* row_sb = (float*)alloc(N_ROWS * 4);
  float* P      = (float*)alloc(N_ROWS * 4);
  float* Q      = (float*)alloc(N_ROWS * 4);
  float* omp    = (float*)alloc(N_ROWS * 4);
  float* pcp    = (float*)alloc(N_ROWS * 4);
  float* rm     = (float*)alloc(N_ROWS * 4);
  float* rs     = (float*)alloc(N_ROWS * 4);
  float* b2     = (float*)alloc(4);

  k_f32_to_bf16<<<4096, 256, 0, stream>>>(hidden, hidB, N_ROWS * DIM);
  k_f32_to_bf16<<<16384, 256, 0, stream>>>(W, WB, VOCAB * DIM);
  k_b2sum<<<1, 256, 0, stream>>>(b, b2);
  k_gemm_wmma<<<dim3(VOCAB / 128, N_ROWS / 128), 256, 0, stream>>>(hidB, WB, out);
  k_rowred<<<N_ROWS, 256, 0, stream>>>(out, b, row_s2, row_sb);
  k_scalars<<<N_ROWS, 256, 0, stream>>>(hidden, Wc, bc, b2, row_s2, row_sb, P, Q, omp, pcp);
  k_softstats<<<N_ROWS, 256, 0, stream>>>(out, b, P, Q, pad, rm, rs);
  k_writeprob<<<dim3(VOCAB / 256, N_ROWS), 256, 0, stream>>>(out, b, P, Q, omp, rm, rs, pad);
  k_copygen<<<dim3(8, BATCH), 256, 0, stream>>>(attn, src, pcp, out);
}
